// Model_18726057410614
// MI455X (gfx1250) — matrix operand order repaired and checked by reading, not compiled, not run on hardware
//
#include <hip/hip_runtime.h>
#include <hip/hip_bf16.h>
#include <math.h>

// ---------------- model constants ----------------
#define B_    2
#define S_    4096
#define HID_  768
#define NH_   12
#define HD_   64
#define FFN_  3072
#define NL_   2
#define W_    256
#define G_    1
#define NB_   16
#define TOK_  (B_ * S_)          // 8192
#define SCALE_ 0.125f
#define NEG_  (-1.0e9f)

typedef _Float16 half_t;
typedef half_t v16h __attribute__((ext_vector_type(16)));
typedef half_t v8h  __attribute__((ext_vector_type(8)));
typedef float  v8f  __attribute__((ext_vector_type(8)));

// Build a 16-half WMMA operand from two aligned 8-half groups.
static __device__ inline v16h ld16(const half_t* p0, const half_t* p1) {
  v8h lo = *(const v8h*)p0;
  v8h hi = *(const v8h*)p1;
  return __builtin_shufflevector(lo, hi, 0,1,2,3,4,5,6,7,8,9,10,11,12,13,14,15);
}

static __device__ inline v8f wmma_f16(v16h a, v16h b, v8f c) {
  // (neg_a, A, neg_b, B, c_mod, C, reuse_a, reuse_b)
  return __builtin_amdgcn_wmma_f32_16x16x32_f16(false, a, false, b, (short)0, c, false, false);
}

static __device__ inline float block_sum(float v, float* red) {
  int tid = threadIdx.x;
  red[tid] = v; __syncthreads();
  #pragma unroll
  for (int s = 128; s > 0; s >>= 1) {
    if (tid < s) red[tid] += red[tid + s];
    __syncthreads();
  }
  float r = red[0]; __syncthreads();
  return r;
}

// ---------------- embedding + layernorm ----------------
__global__ __launch_bounds__(256) void embed_ln_kernel(
    const int* __restrict__ ids, const float* __restrict__ word,
    const float* __restrict__ pos, const float* __restrict__ typ,
    const float* __restrict__ g, const float* __restrict__ bta,
    float* __restrict__ x, half_t* __restrict__ xh) {
  __shared__ float red[256];
  const int t = blockIdx.x;           // token 0..TOK-1
  const int s = t & (S_ - 1);
  const int tid = threadIdx.x;
  const size_t base = (size_t)t * HID_;
  const int id = ids[t];
  float v[3];
  #pragma unroll
  for (int i = 0; i < 3; ++i) {
    int j = tid + i * 256;
    v[i] = word[(size_t)id * HID_ + j] + pos[(size_t)s * HID_ + j] + typ[j];
  }
  float mean = block_sum(v[0] + v[1] + v[2], red) * (1.0f / HID_);
  float q = 0.f;
  #pragma unroll
  for (int i = 0; i < 3; ++i) { float d = v[i] - mean; q += d * d; }
  float var = block_sum(q, red) * (1.0f / HID_);
  float r = rsqrtf(var + 1e-5f);
  #pragma unroll
  for (int i = 0; i < 3; ++i) {
    int j = tid + i * 256;
    float y = (v[i] - mean) * r * g[j] + bta[j];
    x[base + j] = y;
    xh[base + j] = (half_t)y;
  }
}

// ---------------- residual + layernorm ----------------
__global__ __launch_bounds__(256) void ln_add_kernel(
    const float* __restrict__ xin, const float* __restrict__ tmp,
    const float* __restrict__ g, const float* __restrict__ bta,
    float* __restrict__ xout, half_t* __restrict__ xh) {
  __shared__ float red[256];
  const int t = blockIdx.x;
  const int tid = threadIdx.x;
  const size_t base = (size_t)t * HID_;
  float v[3];
  #pragma unroll
  for (int i = 0; i < 3; ++i) {
    int j = tid + i * 256;
    v[i] = xin[base + j] + tmp[base + j];
  }
  float mean = block_sum(v[0] + v[1] + v[2], red) * (1.0f / HID_);
  float q = 0.f;
  #pragma unroll
  for (int i = 0; i < 3; ++i) { float d = v[i] - mean; q += d * d; }
  float var = block_sum(q, red) * (1.0f / HID_);
  float r = rsqrtf(var + 1e-5f);
  #pragma unroll
  for (int i = 0; i < 3; ++i) {
    int j = tid + i * 256;
    float y = (v[i] - mean) * r * g[j] + bta[j];
    xout[base + j] = y;
    xh[base + j] = (half_t)y;
  }
}

// ---------------- f32 [K][N] -> f16 transposed [N][K] ----------------
__global__ __launch_bounds__(256) void convT_kernel(
    const float* __restrict__ w, half_t* __restrict__ wt, int K, int N) {
  int idx = blockIdx.x * 256 + threadIdx.x;
  if (idx >= K * N) return;
  int k = idx / N, n = idx - k * N;            // coalesced read of w
  wt[(size_t)n * K + k] = (half_t)w[idx];
}

// ---------------- WMMA GEMM: C = A[M,K] * Bt[N,K]^T + bias ----------------
// Double-buffered LDS, K stepped by 64 (16 WMMA per barrier), software-pipelined
// global->LDS staging so load latency overlaps the matrix-pipe work.
// mode 0: f32 out [M][N]    (alpha applied)
// mode 1: f16 head-scatter  out[((b*NH+h)*S + s)*HD + d]  (alpha applied)
// mode 2: f16 GELU out [M][N]
#define LDK 72   // 64 K-halves + 8 pad (144 B stride: 16B-aligned groups, bank-skewed)
__global__ __launch_bounds__(256) void gemm_wmma_kernel(
    const half_t* __restrict__ A, const half_t* __restrict__ Bt,
    const float* __restrict__ bias, void* __restrict__ dst,
    int M, int N, int K, float alpha, int mode) {
  __shared__ __align__(16) half_t As[2][128 * LDK];
  __shared__ __align__(16) half_t Bs[2][128 * LDK];
  const int tid = threadIdx.x;
  const int lane = tid & 31, wv = tid >> 5;
  const int lh = lane >> 4, l15 = lane & 15;
  const int wr = wv >> 2, wc = wv & 3;         // 2x4 wave grid
  const int row0 = blockIdx.y * 128, col0 = blockIdx.x * 128;
  const int arow = tid >> 1, ak = (tid & 1) * 32;   // 2 x uint4 per thread per matrix

  const half_t* ag = A  + (size_t)(row0 + arow) * K + ak;
  const half_t* bg = Bt + (size_t)(col0 + arow) * K + ak;

  v8f acc[4][2];
  #pragma unroll
  for (int mi = 0; mi < 4; ++mi)
    #pragma unroll
    for (int ni = 0; ni < 2; ++ni)
      acc[mi][ni] = (v8f){0.f,0.f,0.f,0.f,0.f,0.f,0.f,0.f};

  // prologue: stage tile 0 into buffer 0
  {
    uint4 a0 = *(const uint4*)(ag);
    uint4 a1 = *(const uint4*)(ag + 16);
    uint4 b0 = *(const uint4*)(bg);
    uint4 b1 = *(const uint4*)(bg + 16);
    *(uint4*)(&As[0][arow * LDK + ak])      = a0;
    *(uint4*)(&As[0][arow * LDK + ak + 16]) = a1;
    *(uint4*)(&Bs[0][arow * LDK + ak])      = b0;
    *(uint4*)(&Bs[0][arow * LDK + ak + 16]) = b1;
  }
  __syncthreads();

  const int nk = K >> 6;                        // K is a multiple of 64
  for (int kt = 0; kt < nk; ++kt) {
    const int cur = kt & 1, nxt = cur ^ 1;

    // stage tile kt+1 into the spare buffer (overlaps with compute below)
    if (kt + 1 < nk) {
      const half_t* an = ag + (size_t)(kt + 1) * 64;
      const half_t* bn = bg + (size_t)(kt + 1) * 64;
      uint4 a0 = *(const uint4*)(an);
      uint4 a1 = *(const uint4*)(an + 16);
      uint4 b0 = *(const uint4*)(bn);
      uint4 b1 = *(const uint4*)(bn + 16);
      *(uint4*)(&As[nxt][arow * LDK + ak])      = a0;
      *(uint4*)(&As[nxt][arow * LDK + ak + 16]) = a1;
      *(uint4*)(&Bs[nxt][arow * LDK + ak])      = b0;
      *(uint4*)(&Bs[nxt][arow * LDK + ak + 16]) = b1;
      if (kt + 2 < nk) {
        __builtin_prefetch(an + 64, 0, 1);
        __builtin_prefetch(bn + 64, 0, 1);
      }
    }

    // 16 WMMA on the current buffer (two 32-wide sub-steps)
    #pragma unroll
    for (int h2 = 0; h2 < 2; ++h2) {
      v16h a[4], b[2];
      #pragma unroll
      for (int mi = 0; mi < 4; ++mi) {
        const half_t* p = &As[cur][(wr * 64 + mi * 16 + l15) * LDK + h2 * 32 + 8 * lh];
        a[mi] = ld16(p, p + 16);      // K pattern {0..7,16..23} / {8..15,24..31}
      }
      #pragma unroll
      for (int ni = 0; ni < 2; ++ni) {
        const half_t* p = &Bs[cur][(wc * 32 + ni * 16 + l15) * LDK + h2 * 32 + 8 * lh];
        b[ni] = ld16(p, p + 16);      // K pattern {0..7,16..23} / {8..15,24..31}
      }
      #pragma unroll
      for (int mi = 0; mi < 4; ++mi)
        #pragma unroll
        for (int ni = 0; ni < 2; ++ni)
          acc[mi][ni] = wmma_f16(a[mi], b[ni], acc[mi][ni]);
    }

    // single barrier per K-step: releases buffer `nxt` for compute and
    // protects buffer `cur` from being overwritten next iteration.
    __syncthreads();
  }

  #pragma unroll
  for (int mi = 0; mi < 4; ++mi) {
    #pragma unroll
    for (int ni = 0; ni < 2; ++ni) {
      int col = col0 + wc * 32 + ni * 16 + l15;
      float bc = bias[col];
      #pragma unroll
      for (int j = 0; j < 8; ++j) {
        int row = row0 + wr * 64 + mi * 16 + j + 8 * lh;
        float v = (acc[mi][ni][j] + bc) * alpha;
        if (mode == 0) {
          ((float*)dst)[(size_t)row * N + col] = v;
        } else if (mode == 1) {
          int bI = row >> 12, s = row & (S_ - 1);
          int h = col >> 6, d = col & 63;
          ((half_t*)dst)[((size_t)(bI * NH_ + h) * S_ + s) * HD_ + d] = (half_t)v;
        } else {
          float ge = 0.5f * v * (1.0f + erff(v * 0.70710678118f));
          ((half_t*)dst)[(size_t)row * N + col] = (half_t)ge;
        }
      }
    }
  }
}

// ---------------- band (sliding-window) attention, flash-style ----------------
__global__ __launch_bounds__(512) void band_attn_kernel(
    const half_t* __restrict__ qh, const half_t* __restrict__ kh,
    const half_t* __restrict__ vh, const int* __restrict__ am,
    half_t* __restrict__ aoh) {
  const int nb = blockIdx.x;   // query block 0..NB-1
  const int h  = blockIdx.y;
  const int b  = blockIdx.z;
  const int tid = threadIdx.x, wv = tid >> 5, lane = tid & 31;
  const int lh = lane >> 4, l15 = lane & 15;

  __shared__ __align__(16) half_t kbuf[32 * 72];        // [kk][d], stride 72
  __shared__ __align__(16) half_t vbuf[64 * 40];        // [d][kk], stride 40
  __shared__ __align__(16) half_t pbuf[16 * 16 * 40];   // [wave][row][k], stride 40
  __shared__ int ambuf[32];

  const size_t hb = (size_t)(b * NH_ + h) * S_ * HD_;

  // A-operand (queries) for this wave: rows wv*16 .. wv*16+15, k split 0..31 / 32..63
  v16h aq0, aq1;
  {
    const half_t* qp = qh + hb + (size_t)(nb * W_ + wv * 16 + l15) * HD_;
    aq0 = ld16(qp + 16 * lh,      qp + 8 + 16 * lh);
    aq1 = ld16(qp + 32 + 16 * lh, qp + 40 + 16 * lh);
  }

  v8f acc[4];
  float m[8], l[8];

  // ---- init online-softmax state with the single global column (token 0) ----
  float sg = 0.f;
  {
    const half_t* qp = qh + hb + (size_t)(nb * W_ + wv * 16 + l15) * HD_;
    const half_t* kp = kh + hb;   // s = 0
    #pragma unroll
    for (int d = 0; d < HD_; ++d) sg += (float)qp[d] * (float)kp[d];
  }
  const int am0 = am[b * S_];
  float v0c[4];
  #pragma unroll
  for (int nt = 0; nt < 4; ++nt) v0c[nt] = (float)vh[hb + nt * 16 + l15];
  #pragma unroll
  for (int j = 0; j < 8; ++j) {
    float a0 = __shfl(sg, j, 32), a1 = __shfl(sg, j + 8, 32);
    float sv = lh ? a1 : a0;
    if (am0 <= 0) sv = NEG_;
    m[j] = sv; l[j] = 1.0f;
  }
  #pragma unroll
  for (int nt = 0; nt < 4; ++nt) {
    v8f t;
    #pragma unroll
    for (int j = 0; j < 8; ++j) t[j] = v0c[nt];   // p == 1 at init
    acc[nt] = t;
  }

  // ---- stream 24 chunks of 32 band keys ----
  for (int c = 0; c < 24; ++c) {
    // cooperative stage of K (row-major) and V (transposed) + attention mask
    {
      int idx = tid * 4;
      int kkr = idx >> 6, d0 = idx & 63;
      int posk = (nb - 1) * W_ + c * 32 + kkr;
      unsigned long long kvq = 0ull, vvq = 0ull;
      if (posk >= 0 && posk < S_) {
        kvq = *(const unsigned long long*)(kh + hb + (size_t)posk * HD_ + d0);
        vvq = *(const unsigned long long*)(vh + hb + (size_t)posk * HD_ + d0);
      }
      *(unsigned long long*)(&kbuf[kkr * 72 + d0]) = kvq;
      union { unsigned long long u; half_t hh[4]; } uv; uv.u = vvq;
      #pragma unroll
      for (int i = 0; i < 4; ++i) vbuf[(d0 + i) * 40 + kkr] = uv.hh[i];
      if (tid < 32) {
        int pk = (nb - 1) * W_ + c * 32 + tid;
        ambuf[tid] = (pk >= 0 && pk < S_) ? am[b * S_ + pk] : 0;
      }
    }
    __syncthreads();

    // scores: two 16-key tiles, each = 2 WMMA over HD=64
    float s0[8], s1[8];
    #pragma unroll
    for (int t = 0; t < 2; ++t) {
      v8f sa = (v8f){0.f,0.f,0.f,0.f,0.f,0.f,0.f,0.f};
      const half_t* kp = &kbuf[(t * 16 + l15) * 72 + 16 * lh];
      v16h b0 = ld16(kp, kp + 8);            // d 0..31
      v16h b1 = ld16(kp + 32, kp + 40);      // d 32..63
      sa = wmma_f16(aq0, b0, sa);
      sa = wmma_f16(aq1, b1, sa);
      int posk = (nb - 1) * W_ + c * 32 + t * 16 + l15;
      bool kval = (posk >= G_) && (posk < S_) && (ambuf[t * 16 + l15] > 0);
      float* so = t ? s1 : s0;
      #pragma unroll
      for (int j = 0; j < 8; ++j) {
        int sq = nb * W_ + wv * 16 + j + 8 * lh;
        int dd = posk - sq;
        bool ok = kval && (dd <= W_) && (dd >= -W_);
        so[j] = ok ? sa[j] : NEG_;
      }
    }

    // online softmax update + stash probabilities (f16) for PV A-operand
    #pragma unroll
    for (int j = 0; j < 8; ++j) {
      float mx = fmaxf(s0[j], s1[j]);
      #pragma unroll
      for (int o = 1; o < 16; o <<= 1) mx = fmaxf(mx, __shfl_xor(mx, o, 32));
      float nm = fmaxf(m[j], mx);
      float sc = __expf(m[j] - nm);
      float p0 = __expf(s0[j] - nm);
      float p1 = __expf(s1[j] - nm);
      float rs = p0 + p1;
      #pragma unroll
      for (int o = 1; o < 16; o <<= 1) rs += __shfl_xor(rs, o, 32);
      l[j] = l[j] * sc + rs;
      m[j] = nm;
      #pragma unroll
      for (int nt = 0; nt < 4; ++nt) acc[nt][j] *= sc;
      int prow = wv * 16 + j + 8 * lh;
      pbuf[prow * 40 + l15]      = (half_t)p0;
      pbuf[prow * 40 + 16 + l15] = (half_t)p1;
    }
    asm volatile("s_wait_dscnt 0x0" ::: "memory");   // wave-local LDS RAW fence

    // P (16x32) @ V (32x64): 4 WMMA
    const half_t* pp = &pbuf[(wv * 16 + l15) * 40];
    v16h ap = ld16(pp + 16 * lh, pp + 8 + 16 * lh);
    #pragma unroll
    for (int nt = 0; nt < 4; ++nt) {
      const half_t* vp = &vbuf[(nt * 16 + l15) * 40 + 16 * lh];
      v16h bv = ld16(vp, vp + 8);
      acc[nt] = wmma_f16(ap, bv, acc[nt]);
    }
    __syncthreads();
  }

  // normalize and store f16 in token-major layout [b,s][h*64+d]
  #pragma unroll
  for (int j = 0; j < 8; ++j) {
    float inv = 1.0f / l[j];
    int sq = nb * W_ + wv * 16 + j + 8 * lh;
    size_t rowb = ((size_t)(b * S_ + sq)) * HID_ + h * HD_;
    #pragma unroll
    for (int nt = 0; nt < 4; ++nt)
      aoh[rowb + nt * 16 + l15] = (half_t)(acc[nt][j] * inv);
  }
}

// ---------------- global-token attention (G rows, full S keys) ----------------
__global__ __launch_bounds__(256) void global_attn_kernel(
    const half_t* __restrict__ qgh, const half_t* __restrict__ kgh,
    const half_t* __restrict__ vgh, const int* __restrict__ am,
    half_t* __restrict__ aoh) {
  const int h = blockIdx.x, b = blockIdx.y;
  __shared__ float sc[S_];
  __shared__ float red[256];
  __shared__ float qv[HD_];
  const size_t hb = (size_t)(b * NH_ + h) * S_ * HD_;
  const int tid = threadIdx.x;
  if (tid < HD_) qv[tid] = (float)qgh[hb + tid];   // global query row 0 (scaled)
  __syncthreads();
  float lmax = -3.0e38f;
  for (int k = tid; k < S_; k += 256) {
    const half_t* kp = kgh + hb + (size_t)k * HD_;
    float s = 0.f;
    #pragma unroll
    for (int d = 0; d < HD_; ++d) s += qv[d] * (float)kp[d];
    if (am[b * S_ + k] <= 0) s = NEG_;
    sc[k] = s;
    lmax = fmaxf(lmax, s);
  }
  red[tid] = lmax; __syncthreads();
  #pragma unroll
  for (int o = 128; o > 0; o >>= 1) { if (tid < o) red[tid] = fmaxf(red[tid], red[tid + o]); __syncthreads(); }
  float gmax = red[0]; __syncthreads();
  float lsum = 0.f;
  for (int k = tid; k < S_; k += 256) {
    float p = __expf(sc[k] - gmax);
    sc[k] = p; lsum += p;
  }
  float gsum = block_sum(lsum, red);
  float inv = 1.0f / gsum;
  __syncthreads();
  int d = tid & 63, sl = tid >> 6;
  float pa = 0.f;
  for (int k = sl * 1024; k < sl * 1024 + 1024; ++k)
    pa += sc[k] * (float)vgh[hb + (size_t)k * HD_ + d];
  red[tid] = pa; __syncthreads();
  if (tid < 64) {
    float o = (red[tid] + red[tid + 64] + red[tid + 128] + red[tid + 192]) * inv;
    aoh[((size_t)(b * S_)) * HID_ + h * HD_ + tid] = (half_t)o;   // row s=0
  }
}

// ---------------- pooler (tanh(x[:,0] @ pool_w + b)) ----------------
__global__ __launch_bounds__(256) void pool_kernel(
    const float* __restrict__ x, const float* __restrict__ pw,
    const float* __restrict__ pb, float* __restrict__ pooled) {
  const int b = blockIdx.x;
  const float* xr = x + (size_t)b * S_ * HID_;   // token (b, 0)
  for (int j = threadIdx.x; j < HID_; j += 256) {
    float a = pb[j];
    for (int k = 0; k < HID_; ++k) a += xr[k] * pw[(size_t)k * HID_ + j];
    pooled[b * HID_ + j] = tanhf(a);
  }
}

// ---------------- classifier ----------------
__global__ __launch_bounds__(32) void cls_kernel(
    const float* __restrict__ pooled, const float* __restrict__ cw,
    const float* __restrict__ cb, float* __restrict__ out) {
  int i = threadIdx.x;
  if (i < B_ * 10) {
    int b = i / 10, c = i - b * 10;
    float a = cb[c];
    for (int k = 0; k < HID_; ++k) a += pooled[b * HID_ + k] * cw[(size_t)k * 10 + c];
    out[i] = a;
  }
}

// ==================== host orchestration ====================
extern "C" void kernel_launch(void* const* d_in, const int* in_sizes, int n_in,
                              void* d_out, int out_size, void* d_ws, size_t ws_size,
                              hipStream_t stream) {
  (void)in_sizes; (void)n_in; (void)out_size; (void)ws_size;
  const int* ids   = (const int*)d_in[0];
  const int* amask = (const int*)d_in[1];
  const float* word = (const float*)d_in[2];
  const float* pos  = (const float*)d_in[3];
  const float* typ  = (const float*)d_in[4];
  const float* eg   = (const float*)d_in[5];
  const float* eb   = (const float*)d_in[6];
  const float* pool_w = (const float*)d_in[51];
  const float* pool_b = (const float*)d_in[52];
  const float* cls_w  = (const float*)d_in[53];
  const float* cls_b  = (const float*)d_in[54];

  // ---- workspace carve-up (all chunks 256B-aligned by construction) ----
  char* ws = (char*)d_ws;
  float*  x    = (float*)ws;  ws += (size_t)TOK_ * HID_ * 4;
  half_t* xh   = (half_t*)ws; ws += (size_t)TOK_ * HID_ * 2;
  float*  tmp  = (float*)ws;  ws += (size_t)TOK_ * HID_ * 4;
  half_t* wt   = (half_t*)ws; ws += (size_t)FFN_ * HID_ * 2;   // max weight (768x3072)
  half_t* qhh  = (half_t*)ws; ws += (size_t)TOK_ * HID_ * 2;
  half_t* khh  = (half_t*)ws; ws += (size_t)TOK_ * HID_ * 2;
  half_t* vhh  = (half_t*)ws; ws += (size_t)TOK_ * HID_ * 2;
  half_t* qgh  = (half_t*)ws; ws += (size_t)TOK_ * HID_ * 2;
  half_t* kgh  = (half_t*)ws; ws += (size_t)TOK_ * HID_ * 2;
  half_t* vgh  = (half_t*)ws; ws += (size_t)TOK_ * HID_ * 2;
  half_t* aoh  = (half_t*)ws; ws += (size_t)TOK_ * HID_ * 2;
  half_t* h1h  = (half_t*)ws; ws += (size_t)TOK_ * FFN_ * 2;
  float*  pooled = (float*)ws; ws += (size_t)B_ * HID_ * 4;

  auto convT = [&](const float* w, int K, int N) {
    int total = K * N;
    convT_kernel<<<(total + 255) / 256, 256, 0, stream>>>(w, wt, K, N);
  };
  auto gemm = [&](const half_t* A, const float* bias, void* dst,
                  int M, int N, int K, float alpha, int mode) {
    dim3 grid(N / 128, M / 128);
    gemm_wmma_kernel<<<grid, 256, 0, stream>>>(A, wt, bias, dst, M, N, K, alpha, mode);
  };

  embed_ln_kernel<<<TOK_, 256, 0, stream>>>(ids, word, pos, typ, eg, eb, x, xh);

  for (int L = 0; L < NL_; ++L) {
    const int pi = 7 + 22 * L;
    auto P = [&](int i) { return (const float*)d_in[pi + i]; };

    // q,k,v,qg,kg,vg projections -> head-layout f16 (SCALE folded into q/qg)
    struct { int w, b; float a; half_t* dst; } proj[6] = {
      {0, 1, SCALE_, qhh}, {2, 3, 1.0f, khh}, {4, 5, 1.0f, vhh},
      {6, 7, SCALE_, qgh}, {8, 9, 1.0f, kgh}, {10, 11, 1.0f, vgh},
    };
    for (int i = 0; i < 6; ++i) {
      convT(P(proj[i].w), HID_, HID_);
      gemm(xh, P(proj[i].b), proj[i].dst, TOK_, HID_, HID_, proj[i].a, /*heads*/1);
    }

    band_attn_kernel<<<dim3(NB_, NH_, B_), 512, 0, stream>>>(qhh, khh, vhh, amask, aoh);
    global_attn_kernel<<<dim3(NH_, B_), 256, 0, stream>>>(qgh, kgh, vgh, amask, aoh);

    // output projection + residual LN1
    convT(P(12), HID_, HID_);
    gemm(aoh, P(13), tmp, TOK_, HID_, HID_, 1.0f, /*f32*/0);
    ln_add_kernel<<<TOK_, 256, 0, stream>>>(x, tmp, P(14), P(15), x, xh);

    // FFN
    convT(P(16), HID_, FFN_);
    gemm(xh, P(17), h1h, TOK_, FFN_, HID_, 1.0f, /*gelu f16*/2);
    convT(P(18), FFN_, HID_);
    gemm(h1h, P(19), tmp, TOK_, HID_, FFN_, 1.0f, /*f32*/0);
    ln_add_kernel<<<TOK_, 256, 0, stream>>>(x, tmp, P(20), P(21), x, xh);
  }

  pool_kernel<<<B_, 256, 0, stream>>>(x, pool_w, pool_b, pooled);
  cls_kernel<<<1, 32, 0, stream>>>(pooled, cls_w, cls_b, (float*)d_out);
}
